// BackwardDecoder_45767171506730
// MI455X (gfx1250) — matrix operand order repaired and checked by reading, not compiled, not run on hardware
//
#include <hip/hip_runtime.h>
#include <hip/hip_bf16.h>
#include <math.h>

typedef __attribute__((ext_vector_type(16))) __bf16 v16bf;
typedef __attribute__((ext_vector_type(8)))  __bf16 v8bf;
typedef __attribute__((ext_vector_type(8)))  float  v8f;

static constexpr int Bn = 32, Tn = 64, Sn = 64, Vn = 32000;
static constexpr int En = 512, Hn = 512, Un = 1024, NHn = 8, Dn = 64, DVn = 128;

// ---------------- WMMA fragment loaders (contiguous k = 16*(lane/16) + s in A and B) ----------------
struct A2 { v8bf lo, hi; };

__device__ __forceinline__ v16bf load_a_frag(const __bf16* A, int lda, int m0, int k0, int lane) {
  // 16-bit A 16x32: lanes 0-15 rows M=0..15 with K 0..15; lanes 16-31 K 16..31
  int half = lane >> 4, row = m0 + (lane & 15);
  const __bf16* p = A + (size_t)row * lda + k0 + half * 16;
  A2 t;
  t.lo = *(const v8bf*)(p);
  t.hi = *(const v8bf*)(p + 8);
  return __builtin_bit_cast(v16bf, t);
}

__device__ __forceinline__ v16bf load_a_frag(const float* A, int lda, int m0, int k0, int lane) {
  int half = lane >> 4, row = m0 + (lane & 15);
  const float* p = A + (size_t)row * lda + k0 + half * 16;
  v16bf a;
#pragma unroll
  for (int i = 0; i < 8; ++i) a[i] = (__bf16)p[i];
#pragma unroll
  for (int i = 0; i < 8; ++i) a[8 + i] = (__bf16)p[8 + i];
  return a;
}

__device__ __forceinline__ v16bf load_b_frag(const __bf16* W, int K, int n0, int k0, int lane) {
  // B 32x16: lane = column n0+(l&15); elems 0..15 = K (l>>4)*16 .. +15 (contiguous in W[n][k])
  int col = n0 + (lane & 15);
  int kb = k0 + (lane >> 4) * 16;
  return *(const v16bf*)(W + (size_t)col * K + kb);
}

// C = act(A[M,K] * W[N,K]^T + bias). One wave computes a 16 x (16*NT) strip:
// A fragment loaded once per K-step, reused across NT B fragments (register blocking).
template <typename TA, typename TO, int NT>
__global__ void __launch_bounds__(256) k_gemm(const TA* __restrict__ A, int lda,
                                              const __bf16* __restrict__ W,
                                              const float* __restrict__ bias,
                                              TO* __restrict__ C, int ldc,
                                              int M, int N, int K, int act) {
  const int lane = threadIdx.x & 31;
  const int wave = threadIdx.x >> 5;
  const int m0 = blockIdx.x * 16;
  const int n0 = (blockIdx.y * 8 + wave) * (16 * NT);
  if (m0 >= M || n0 >= N) return;  // N % (16*NT) == 0 for all call sites
  v8f acc[NT];
#pragma unroll
  for (int j = 0; j < NT; ++j) acc[j] = (v8f){};
  for (int k0 = 0; k0 < K; k0 += 32) {
    v16bf a = load_a_frag(A, lda, m0, k0, lane);
    __builtin_prefetch(W + (size_t)(n0 + (lane & 15)) * K + k0 + 64, 0, 1);
#pragma unroll
    for (int j = 0; j < NT; ++j) {
      v16bf b = load_b_frag(W, K, n0 + 16 * j, k0, lane);
      acc[j] = __builtin_amdgcn_wmma_f32_16x16x32_bf16(false, a, false, b, (short)0,
                                                       acc[j], false, false);
    }
  }
  const int r0 = m0 + (lane >> 4) * 8;
#pragma unroll
  for (int j = 0; j < NT; ++j) {
    const int col = n0 + 16 * j + (lane & 15);
    const float bv = bias ? bias[col] : 0.0f;
#pragma unroll
    for (int v = 0; v < 8; ++v) {
      float x = acc[j][v] + bv;
      if (act == 1) x = tanhf(x);
      C[(size_t)(r0 + v) * ldc + col] = (TO)x;
    }
  }
}

// ---------------- elementwise / small kernels ----------------
__global__ void k_f2bf(const float* __restrict__ s, __bf16* __restrict__ d, size_t n) {
  size_t i = (size_t)blockIdx.x * blockDim.x + threadIdx.x;
  size_t st = (size_t)gridDim.x * blockDim.x;
  for (; i < n; i += st) d[i] = (__bf16)s[i];
}

__global__ void k_embed(const int* __restrict__ tok, const float* __restrict__ ew,
                        float* __restrict__ emb) {
  size_t i = (size_t)blockIdx.x * blockDim.x + threadIdx.x;
  size_t n = (size_t)Bn * Tn * En;
  if (i >= n) return;
  int e = (int)(i % En);
  int bt = (int)(i / En);
  emb[i] = ew[(size_t)tok[bt] * En + e];
}

// enc_out is [S,B,U]; produce bf16 enc[b*S+s][u]
__global__ void k_encbf(const float* __restrict__ enc_out, __bf16* __restrict__ encbf) {
  size_t i = (size_t)blockIdx.x * blockDim.x + threadIdx.x;
  size_t n = (size_t)Bn * Sn * Un;
  if (i >= n) return;
  int u = (int)(i % Un);
  int bs = (int)(i / Un);
  int s = bs % Sn, b = bs / Sn;
  encbf[i] = (__bf16)enc_out[(size_t)s * Bn * Un + (size_t)b * Un + u];
}

__global__ void __launch_bounds__(256) k_bridge(const float* __restrict__ enc_out,
                                                const unsigned char* __restrict__ mask,
                                                const float* __restrict__ bW,
                                                const float* __restrict__ bb,
                                                float* __restrict__ h0) {
  int b = blockIdx.x;
  __shared__ int sl;
  if (threadIdx.x == 0) {
    int cnt = 0;
    for (int s = 0; s < Sn; ++s) cnt += mask[b * Sn + s] ? 1 : 0;
    sl = Sn - cnt;
  }
  __syncthreads();
  int len = sl;
  const float* fn = enc_out + (size_t)(len - 1) * Bn * Un + (size_t)b * Un;  // fwd half
  for (int i = threadIdx.x; i < Hn; i += blockDim.x) {
    float acc = bb[i];
    const float* w = bW + (size_t)i * (Un / 2);
    for (int j = 0; j < Un / 2; ++j) acc += fn[j] * w[j];
    h0[(size_t)b * Hn + i] = tanhf(acc);
  }
}

__device__ __forceinline__ float sigf(float x) { return 1.0f / (1.0f + expf(-x)); }

__global__ void __launch_bounds__(256) k_gru(const float* __restrict__ gx,
                                             const float* __restrict__ gh,
                                             const float* __restrict__ hprev,
                                             float* __restrict__ o0, int ld0,
                                             float* __restrict__ o1, int ld1, int Hd) {
  int b = blockIdx.x;
  const float* gxr = gx + (size_t)b * 3 * Hd;
  const float* ghr = gh + (size_t)b * 3 * Hd;
  for (int i = threadIdx.x; i < Hd; i += blockDim.x) {
    float r = sigf(gxr[i] + ghr[i]);
    float z = sigf(gxr[Hd + i] + ghr[Hd + i]);
    float nn = tanhf(gxr[2 * Hd + i] + r * ghr[2 * Hd + i]);
    float hv = (1.0f - z) * nn + z * hprev[(size_t)b * Hd + i];
    o0[(size_t)b * ld0 + i] = hv;
    if (o1) o1[(size_t)b * ld1 + i] = hv;
  }
}

// scores -> softmax -> context for one (b, head). block = 128 threads.
__global__ void __launch_bounds__(128) k_attn(const float* __restrict__ q,
                                              const float* __restrict__ keylin,
                                              const float* __restrict__ enc_out,
                                              const unsigned char* __restrict__ mask,
                                              const float* __restrict__ Ww,
                                              const float* __restrict__ bw,
                                              float* __restrict__ ctx) {
  int b = blockIdx.x, hh = blockIdx.y;
  int tid = threadIdx.x;
  __shared__ float sc[Sn];
  __shared__ float inv_s;
  if (tid < Sn) {
    int s = tid;
    const float* kp = keylin + ((size_t)(b * Sn + s)) * (NHn * Dn) + hh * Dn;
    const float* qp = q + (size_t)b * (NHn * Dn) + hh * Dn;
    float acc = 0.0f;
    for (int d = 0; d < Dn; ++d) acc += tanhf(qp[d] + kp[d]) * Ww[d];
    acc += bw[0];
    if (mask[b * Sn + s]) acc += -1e9f;
    sc[s] = acc;
  }
  __syncthreads();
  if (tid == 0) {
    float m = sc[0];
    for (int s = 1; s < Sn; ++s) m = fmaxf(m, sc[s]);
    float sum = 0.0f;
    for (int s = 0; s < Sn; ++s) { sc[s] = expf(sc[s] - m); sum += sc[s]; }
    inv_s = 1.0f / sum;
  }
  __syncthreads();
  float inv = inv_s;
  int v = tid;  // 0..127 = DVn
  float a = 0.0f;
  for (int s = 0; s < Sn; ++s) {
    float val = enc_out[(size_t)s * Bn * Un + (size_t)b * Un + hh * DVn + v];
    a += sc[s] * inv * val;
  }
  ctx[(size_t)b * Un + hh * DVn + v] = a;
}

// feat[b*T+t][0:512]=emb, [512:1024]=hs, [1024:2048]=ctxs  -> bf16
__global__ void k_feat(const float* __restrict__ emb, const float* __restrict__ hs,
                       const float* __restrict__ ctxs, __bf16* __restrict__ feat) {
  size_t i = (size_t)blockIdx.x * blockDim.x + threadIdx.x;
  size_t n = (size_t)Bn * Tn * (En + Hn + Un);
  if (i >= n) return;
  int c = (int)(i & 2047);
  int r = (int)(i >> 11);
  int b = r / Tn, t = r % Tn;
  float v;
  if (c < En) v = emb[((size_t)b * Tn + t) * En + c];
  else if (c < En + Hn) v = hs[((size_t)b * Tn + t) * Hn + (c - En)];
  else v = ctxs[((size_t)b * Tn + t) * Un + (c - En - Hn)];
  feat[i] = (__bf16)v;
}

// ---------------- host launcher ----------------
extern "C" void kernel_launch(void* const* d_in, const int* in_sizes, int n_in,
                              void* d_out, int out_size, void* d_ws, size_t ws_size,
                              hipStream_t stream) {
  (void)in_sizes; (void)n_in; (void)out_size; (void)ws_size;
  const int* tokens = (const int*)d_in[0];
  const unsigned char* enc_mask = (const unsigned char*)d_in[1];
  const float* enc_out = (const float*)d_in[2];
  const float* embed_w = (const float*)d_in[3];
  const float* g1Wx = (const float*)d_in[4];
  const float* g1Wh = (const float*)d_in[5];
  const float* g1bx = (const float*)d_in[6];
  const float* g1bh = (const float*)d_in[7];
  const float* g2Wx = (const float*)d_in[8];
  const float* g2Wh = (const float*)d_in[9];
  const float* g2bx = (const float*)d_in[10];
  const float* g2bh = (const float*)d_in[11];
  const float* bridge_W = (const float*)d_in[12];
  const float* bridge_b = (const float*)d_in[13];
  const float* Wk = (const float*)d_in[14];
  const float* bk = (const float*)d_in[15];
  const float* Wq = (const float*)d_in[16];
  const float* bq = (const float*)d_in[17];
  const float* Ww = (const float*)d_in[18];
  const float* bw = (const float*)d_in[19];
  const float* Wf = (const float*)d_in[20];
  const float* bfv = (const float*)d_in[21];
  const float* Wo = (const float*)d_in[22];
  const float* bo = (const float*)d_in[23];
  float* out = (float*)d_out;

  // bump allocator over workspace
  char* p = (char*)d_ws;
  auto alloc = [&](size_t bytes) -> void* {
    void* r = (void*)p;
    p += (bytes + 255) & ~(size_t)255;
    return r;
  };
  float* emb    = (float*)alloc((size_t)Bn * Tn * En * 4);
  __bf16* encbf = (__bf16*)alloc((size_t)Bn * Sn * Un * 2);
  float* keylin = (float*)alloc((size_t)Bn * Sn * NHn * Dn * 4);
  float* hs     = (float*)alloc((size_t)Bn * Tn * Hn * 4);
  float* ctxs   = (float*)alloc((size_t)Bn * Tn * Un * 4);
  float* h      = (float*)alloc((size_t)Bn * Hn * 4);
  float* tmp    = (float*)alloc((size_t)Bn * Hn * 4);
  float* gxa    = (float*)alloc((size_t)Bn * 3 * Hn * 4);
  float* gxb    = (float*)alloc((size_t)Bn * 3 * Hn * 4);
  float* qbuf   = (float*)alloc((size_t)Bn * NHn * Dn * 4);
  float* ctx    = (float*)alloc((size_t)Bn * Un * 4);
  __bf16* featbf = (__bf16*)alloc((size_t)Bn * Tn * (En + Hn + Un) * 2);
  __bf16* actbf  = (__bf16*)alloc((size_t)Bn * Tn * En * 2);
  __bf16* g1Wx_b = (__bf16*)alloc((size_t)3 * Hn * En * 2);
  __bf16* g1Wh_b = (__bf16*)alloc((size_t)3 * Hn * Hn * 2);
  __bf16* g2Wx_b = (__bf16*)alloc((size_t)3 * Hn * Un * 2);
  __bf16* g2Wh_b = (__bf16*)alloc((size_t)3 * Hn * Hn * 2);
  __bf16* Wq_b   = (__bf16*)alloc((size_t)NHn * Dn * Hn * 2);
  __bf16* Wf_b   = (__bf16*)alloc((size_t)Un * Un * 2);
  __bf16* Wk_b   = (__bf16*)alloc((size_t)NHn * Dn * Un * 2);
  __bf16* Wo_b   = (__bf16*)alloc((size_t)En * (En + Hn + Un) * 2);
  __bf16* emb_b  = (__bf16*)alloc((size_t)Vn * En * 2);

  auto conv = [&](const float* s, __bf16* d, size_t n) {
    int nb = (int)((n + 255) / 256);
    k_f2bf<<<nb, 256, 0, stream>>>(s, d, n);
  };
  constexpr int NT = 4;  // 16x64 strip per wave
  auto gemmFF = [&](const float* A, int lda, const __bf16* W, const float* bias,
                    float* C, int ldc, int M, int N, int K, int act) {
    dim3 g(M / 16, (N / (16 * NT) + 7) / 8);
    k_gemm<float, float, NT><<<g, 256, 0, stream>>>(A, lda, W, bias, C, ldc, M, N, K, act);
  };
  auto gemmBF = [&](const __bf16* A, int lda, const __bf16* W, const float* bias,
                    float* C, int ldc, int M, int N, int K, int act) {
    dim3 g(M / 16, (N / (16 * NT) + 7) / 8);
    k_gemm<__bf16, float, NT><<<g, 256, 0, stream>>>(A, lda, W, bias, C, ldc, M, N, K, act);
  };
  auto gemmBB = [&](const __bf16* A, int lda, const __bf16* W, const float* bias,
                    __bf16* C, int ldc, int M, int N, int K, int act) {
    dim3 g(M / 16, (N / (16 * NT) + 7) / 8);
    k_gemm<__bf16, __bf16, NT><<<g, 256, 0, stream>>>(A, lda, W, bias, C, ldc, M, N, K, act);
  };

  // --- one-time conversions / precompute ---
  conv(g1Wx, g1Wx_b, (size_t)3 * Hn * En);
  conv(g1Wh, g1Wh_b, (size_t)3 * Hn * Hn);
  conv(g2Wx, g2Wx_b, (size_t)3 * Hn * Un);
  conv(g2Wh, g2Wh_b, (size_t)3 * Hn * Hn);
  conv(Wq, Wq_b, (size_t)NHn * Dn * Hn);
  conv(Wf, Wf_b, (size_t)Un * Un);
  conv(Wk, Wk_b, (size_t)NHn * Dn * Un);
  conv(Wo, Wo_b, (size_t)En * (En + Hn + Un));
  conv(embed_w, emb_b, (size_t)Vn * En);

  {
    size_t n = (size_t)Bn * Tn * En;
    k_embed<<<(int)((n + 255) / 256), 256, 0, stream>>>(tokens, embed_w, emb);
  }
  {
    size_t n = (size_t)Bn * Sn * Un;
    k_encbf<<<(int)((n + 255) / 256), 256, 0, stream>>>(enc_out, encbf);
  }
  k_bridge<<<Bn, 256, 0, stream>>>(enc_out, enc_mask, bridge_W, bridge_b, h);
  // key cache: keylin[b*S+s][h*D+d] = enc @ Wk^T + bk
  gemmBF(encbf, Un, Wk_b, bk, keylin, NHn * Dn, Bn * Sn, NHn * Dn, Un, 0);

  // --- sequential decode ---
  for (int t = 0; t < Tn; ++t) {
    // GRU1: x_t = emb[:, t, :]
    gemmFF(emb + (size_t)t * En, Tn * En, g1Wx_b, g1bx, gxa, 3 * Hn, Bn, 3 * Hn, En, 0);
    gemmFF(h, Hn, g1Wh_b, g1bh, gxb, 3 * Hn, Bn, 3 * Hn, Hn, 0);
    k_gru<<<Bn, 256, 0, stream>>>(gxa, gxb, h, tmp, Hn, nullptr, 0, Hn);
    // query
    gemmFF(tmp, Hn, Wq_b, bq, qbuf, NHn * Dn, Bn, NHn * Dn, Hn, 0);
    // additive attention + context
    k_attn<<<dim3(Bn, NHn), 128, 0, stream>>>(qbuf, keylin, enc_out, enc_mask, Ww, bw, ctx);
    // final_linear -> ctxs[:, t, :]
    gemmFF(ctx, Un, Wf_b, bfv, ctxs + (size_t)t * Un, Tn * Un, Bn, Un, Un, 0);
    // GRU2
    gemmFF(ctxs + (size_t)t * Un, Tn * Un, g2Wx_b, g2bx, gxa, 3 * Hn, Bn, 3 * Hn, Un, 0);
    gemmFF(tmp, Hn, g2Wh_b, g2bh, gxb, 3 * Hn, Bn, 3 * Hn, Hn, 0);
    k_gru<<<Bn, 256, 0, stream>>>(gxa, gxb, tmp, h, Hn, hs + (size_t)t * Hn, Tn * Hn, Hn);
  }

  // --- output head ---
  {
    size_t n = (size_t)Bn * Tn * (En + Hn + Un);
    k_feat<<<(int)((n + 255) / 256), 256, 0, stream>>>(emb, hs, ctxs, featbf);
  }
  // act = tanh(feat @ Wo^T + bo)  [2048, 512] bf16
  gemmBB(featbf, En + Hn + Un, Wo_b, bo, actbf, En, Bn * Tn, En, En + Hn + Un, 1);
  // logits = act @ embed_w^T  [2048, 32000] f32 -> d_out
  gemmBF(actbf, En, emb_b, nullptr, out, Vn, Bn * Tn, Vn, En, 0);
}
